// DensityEstimator_17428977287952
// MI455X (gfx1250) — matrix operand order repaired and checked by reading, not compiled, not run on hardware
//
#include <hip/hip_runtime.h>
#include <hip/hip_bf16.h>

// ---------------------------------------------------------------------------
// Neural spline flow log_prob for MI455X (gfx1250, wave32, WMMA).
// GEMMs: bf16 operands pre-swizzled so each wave issues only
//   2x b128 A-loads + 4x b128 B-loads + 2x v_wmma_f32_16x16x32_bf16 per k-step
// (no LDS staging, no exec-masked scalar loads). Weights converted/transposed
// once per layer (<3MB, L2 resident); activations produced directly in bf16
// by producer epilogues. Workspace requirement: ~61 MB.
// ---------------------------------------------------------------------------

typedef __bf16          v16bf __attribute__((ext_vector_type(16)));
typedef unsigned short  u16;
typedef u16             v8us  __attribute__((ext_vector_type(8)));
typedef u16             v16us __attribute__((ext_vector_type(16)));
typedef float           v8f   __attribute__((ext_vector_type(8)));

#define DD      550
#define DKP     576            // DD padded to mult of 32
#define FF      275
#define FKP     288            // FF padded to mult of 32
#define HIDD    128
#define OUTW    6325           // FF * 23
#define OUTP    6336           // OUTW padded to mult of 32
#define BROWS   4096
#define CHUNK   1024
#define TBF     3.0f
#define MINBF   0.001f
#define MINDF   0.001f
#define RSQH    0.08838834764831845f   // 1/sqrt(128)
#define LN2PI   1.8378770664093453f

__device__ __forceinline__ u16 f2bf(float f) {
  unsigned int u = __builtin_bit_cast(unsigned int, f);
  u += 0x7FFFu + ((u >> 16) & 1u);           // round-to-nearest-even
  return (u16)(u >> 16);
}

__device__ __forceinline__ float softplusf(float x) {
  return (x > 20.f) ? x : __logf(1.f + __expf(x));
}

// ------------------------- WMMA GEMM ---------------------------------------
// C[rows x NP] = A_bf[rows x KP] * Wt_bf[NP x KP]^T + bias (+ res)
// A row-major bf16 (lda=KP, zero padded); Wt is W transposed: Wt[n][k], bf16.
// One wave -> 16x32 output tile (two 16x16 WMMA accumulators sharing A).
// rows are exact multiples of 16; KP,NP multiples of 32; only the bias read
// needs a real-N guard.
__global__ void __launch_bounds__(32)
gemm_bf16_wmma(const u16* __restrict__ A, const u16* __restrict__ Wt,
               const float* __restrict__ bias, const float* __restrict__ res,
               float* __restrict__ Cf, u16* __restrict__ Cbf,
               int KP, int Nreal, int NP, int reluOut) {
  const int lane = threadIdx.x & 31;
  const int half = lane >> 4;
  const int ml   = lane & 15;
  const int rowBase = blockIdx.y * 16;
  const int colBase = blockIdx.x * 32;

  const u16* aRow = A + (size_t)(rowBase + ml) * KP + half * 16;
  const u16* b0   = Wt + (size_t)(colBase + ml) * KP + half * 16;
  const u16* b1   = b0 + (size_t)16 * KP;

  v8f acc0 = {}, acc1 = {};
  #pragma unroll 2
  for (int k0 = 0; k0 < KP; k0 += 32) {
    union { v16us v; v8us h[2]; } au, bu0, bu1;
    au.h[0]  = *(const v8us*)(aRow + k0);        // K = 16*half + 0..7
    au.h[1]  = *(const v8us*)(aRow + k0 + 8);    // K = 16*half + 8..15
    bu0.h[0] = *(const v8us*)(b0 + k0);          // K = 16*half + 0..7
    bu0.h[1] = *(const v8us*)(b0 + k0 + 8);      // K = 16*half + 8..15
    bu1.h[0] = *(const v8us*)(b1 + k0);
    bu1.h[1] = *(const v8us*)(b1 + k0 + 8);
    acc0 = __builtin_amdgcn_wmma_f32_16x16x32_bf16(
        false, __builtin_bit_cast(v16bf, au.v),
        false, __builtin_bit_cast(v16bf, bu0.v), (short)0, acc0, false, false);
    acc1 = __builtin_amdgcn_wmma_f32_16x16x32_bf16(
        false, __builtin_bit_cast(v16bf, au.v),
        false, __builtin_bit_cast(v16bf, bu1.v), (short)0, acc1, false, false);
  }

  union { v8f v; float f[8]; } ac[2];
  ac[0].v = acc0; ac[1].v = acc1;
  #pragma unroll
  for (int ct = 0; ct < 2; ++ct) {
    const int col = colBase + ct * 16 + ml;
    const float bz = (col < Nreal) ? bias[col] : 0.f;
    #pragma unroll
    for (int r = 0; r < 8; ++r) {
      const int row = rowBase + (half << 3) + r;   // D layout: m = r + 8*half
      float o = ac[ct].f[r] + bz;
      if (res) o += res[(size_t)row * NP + col];
      if (Cf)  Cf[(size_t)row * NP + col] = o;
      if (Cbf) Cbf[(size_t)row * NP + col] = f2bf(reluOut ? fmaxf(o, 0.f) : o);
    }
  }
}

// ------------------------- weight conversion -------------------------------
// Wt[n*KP + k] = bf16(W[k*N + n]) with zero padding.
__global__ void conv_w_t(const float* __restrict__ W, u16* __restrict__ Wt,
                         int K, int N, int KP, int NP) {
  const int idx = blockIdx.x * blockDim.x + threadIdx.x;
  if (idx >= NP * KP) return;
  const int n = idx / KP, k = idx % KP;
  Wt[idx] = (k < K && n < N) ? f2bf(W[(size_t)k * N + n]) : (u16)0;
}

// M[k,r] = sum_{j<=min(k,r)} U[j,k]*L[r,j];  stored transposed+bf16:
// Mt[r*DKP + k], so it is directly the B operand for z_out = zp @ M.
__global__ void build_Mt(const float* __restrict__ lo, const float* __restrict__ up,
                         const float* __restrict__ udiag, u16* __restrict__ Mt) {
  const int idx = blockIdx.x * blockDim.x + threadIdx.x;
  if (idx >= DKP * DKP) return;
  const int r = idx / DKP, k = idx % DKP;
  if (r >= DD || k >= DD) { Mt[idx] = 0; return; }
  const float diag = softplusf(udiag[k]) + 0.001f;
  const int jm = (k < r) ? k : r;
  float acc = 0.f;
  for (int j = 0; j <= jm; ++j) {
    const float Ujk = (j < k) ? up[(size_t)j * DD + k] : diag;
    const float Lrj = (j < r) ? lo[(size_t)r * DD + j] : 1.f;
    acc = fmaf(Ujk, Lrj, acc);
  }
  Mt[idx] = f2bf(acc);
}

__global__ void gather_perm_bf(const float* __restrict__ src,
                               const int* __restrict__ perm,
                               u16* __restrict__ dst) {
  const int idx = blockIdx.x * blockDim.x + threadIdx.x;
  if (idx >= BROWS * DKP) return;
  const int b = idx / DKP, c = idx % DKP;
  dst[idx] = (c < DD) ? f2bf(src[(size_t)b * DD + perm[c]]) : (u16)0;
}

// ------------------------- RQS spline math ---------------------------------
__device__ void make_knots(const float u[8], float c[9], float wv[8]) {
  float mx = u[0];
  #pragma unroll
  for (int k = 1; k < 8; ++k) mx = fmaxf(mx, u[k]);
  float e[8], sum = 0.f;
  #pragma unroll
  for (int k = 0; k < 8; ++k) { e[k] = __expf(u[k] - mx); sum += e[k]; }
  const float inv = 1.f / sum;
  float cum = 0.f;
  c[0] = -TBF;
  #pragma unroll
  for (int k = 0; k < 8; ++k) {
    cum += MINBF + (1.f - MINBF * 8.f) * e[k] * inv;
    c[k + 1] = 2.f * TBF * cum - TBF;
  }
  c[8] = TBF;
  #pragma unroll
  for (int k = 0; k < 8; ++k) wv[k] = c[k + 1] - c[k];
}

__device__ float rqs_inv(float x, const float* cw, const float* wv,
                         const float* ch, const float* hv, const float* dv,
                         float* ldo) {
  const bool inside = (x >= -TBF) && (x <= TBF);
  const float xi = fminf(fmaxf(x, -TBF), TBF);
  int idx = 0;
  #pragma unroll
  for (int k = 0; k < 9; ++k) {
    const float loc = ch[k] + ((k == 8) ? 1e-6f : 0.f);
    idx += (xi >= loc) ? 1 : 0;
  }
  idx -= 1; idx = (idx < 0) ? 0 : ((idx > 7) ? 7 : idx);
  const float inw = wv[idx], incw = cw[idx], inh = hv[idx], inch = ch[idx];
  const float delta = inh / inw;
  const float d0 = dv[idx], d1 = dv[idx + 1];
  const float t = xi - inch;
  const float s = d0 + d1 - 2.f * delta;
  const float a = t * s + inh * (delta - d0);
  const float b = inh * d0 - t * s;
  const float c = -delta * t;
  const float root = (2.f * c) / (-b - __fsqrt_rn(b * b - 4.f * a * c));
  const float outv = root * inw + incw;
  const float tm = root * (1.f - root);
  const float den = delta + s * tm;
  const float omr = 1.f - root;
  const float dnum = delta * delta * (d1 * root * root + 2.f * delta * tm + d0 * omr * omr);
  const float ld = -(__logf(dnum) - 2.f * __logf(den));
  *ldo = inside ? ld : 0.f;
  return inside ? outv : x;
}

// knot table per feature: [cw:9][w:8][ch:9][h:8][d:9] = 43 floats
__global__ void build_uncond_knots(const float* __restrict__ uw,
                                   const float* __restrict__ uh,
                                   const float* __restrict__ ud,
                                   float* __restrict__ knots) {
  const int f = blockIdx.x * blockDim.x + threadIdx.x;
  if (f >= FF) return;
  float cw[9], wv[8], ch[9], hv[8], dv[9];
  float u0[8], u1[8];
  #pragma unroll
  for (int k = 0; k < 8; ++k) { u0[k] = uw[f * 8 + k]; u1[k] = uh[f * 8 + k]; }
  make_knots(u0, cw, wv);
  make_knots(u1, ch, hv);
  dv[0] = 1.f; dv[8] = 1.f;
  #pragma unroll
  for (int k = 1; k <= 7; ++k) dv[k] = MINDF + softplusf(ud[f * 7 + k - 1]);
  float* kb = knots + (size_t)f * 43;
  #pragma unroll
  for (int k = 0; k < 9; ++k) kb[k] = cw[k];
  #pragma unroll
  for (int k = 0; k < 8; ++k) kb[9 + k] = wv[k];
  #pragma unroll
  for (int k = 0; k < 9; ++k) kb[17 + k] = ch[k];
  #pragma unroll
  for (int k = 0; k < 8; ++k) kb[26 + k] = hv[k];
  #pragma unroll
  for (int k = 0; k < 9; ++k) kb[34 + k] = dv[k];
}

__global__ void uncond_spline(const float* __restrict__ zLin,
                              const float* __restrict__ knots,
                              float* __restrict__ ident, u16* __restrict__ identBf,
                              float* __restrict__ lq) {
  const int row = blockIdx.x;
  const int f = threadIdx.x;
  __shared__ float ssum;
  if (f == 0) ssum = 0.f;
  __syncthreads();
  if (f < FF) {
    const float* kb = knots + (size_t)f * 43;
    float ld;
    const float o = rqs_inv(zLin[(size_t)row * DKP + 2 * f],
                            kb, kb + 9, kb + 17, kb + 26, kb + 34, &ld);
    ident[(size_t)row * FF + f] = o;
    identBf[(size_t)row * FKP + f] = f2bf(o);
    atomicAdd(&ssum, ld);               // ds_add_f32
  } else {
    identBf[(size_t)row * FKP + f] = 0; // zero pad cols FF..FKP-1
  }
  __syncthreads();
  if (f == 0) lq[row] += ssum;
}

__global__ void cond_spline(const float* __restrict__ zLin,
                            const float* __restrict__ ident,
                            const float* __restrict__ p,
                            float* __restrict__ zOut, float* __restrict__ lq,
                            int rowOff) {
  const int row = rowOff + blockIdx.x;
  const int f = threadIdx.x;
  __shared__ float ssum;
  if (f == 0) ssum = 0.f;
  __syncthreads();
  if (f < FF) {
    const float* pp = p + (size_t)blockIdx.x * OUTP + f * 23;
    float uwv[8], uhv[8];
    #pragma unroll
    for (int k = 0; k < 8; ++k) { uwv[k] = pp[k] * RSQH; uhv[k] = pp[8 + k] * RSQH; }
    float cw[9], wv[8], ch[9], hv[8], dv[9];
    make_knots(uwv, cw, wv);
    make_knots(uhv, ch, hv);
    dv[0] = 1.f; dv[8] = 1.f;
    #pragma unroll
    for (int k = 1; k <= 7; ++k) dv[k] = MINDF + softplusf(pp[16 + k - 1]);
    float ld;
    const float tr = rqs_inv(zLin[(size_t)row * DKP + 2 * f + 1],
                             cw, wv, ch, hv, dv, &ld);
    zOut[(size_t)row * DD + 2 * f]     = ident[(size_t)row * FF + f];
    zOut[(size_t)row * DD + 2 * f + 1] = tr;
    atomicAdd(&ssum, ld);
  }
  __syncthreads();
  if (f == 0) lq[row] += ssum;
}

// ------------------------- misc --------------------------------------------
__global__ void ldiag_kernel(const float* __restrict__ udiag, float* __restrict__ out) {
  __shared__ float ssum;
  if (threadIdx.x == 0) ssum = 0.f;
  __syncthreads();
  float v = 0.f;
  for (int c = threadIdx.x; c < DD; c += blockDim.x)
    v += __logf(softplusf(udiag[c]) + 0.001f);
  atomicAdd(&ssum, v);
  __syncthreads();
  if (threadIdx.x == 0) *out = ssum;
}

__global__ void final_kernel(const float* __restrict__ z, const float* __restrict__ loc,
                             const float* __restrict__ lsc, const float* __restrict__ ldg,
                             float* __restrict__ lq) {
  const int row = blockIdx.x;
  __shared__ float ssum;
  if (threadIdx.x == 0) ssum = 0.f;
  __syncthreads();
  float v = 0.f;
  for (int c = threadIdx.x; c < DD; c += blockDim.x) {
    const float ls = lsc[c];
    const float d = (z[(size_t)row * DD + c] - loc[c]) * __expf(-ls);
    v += ls + 0.5f * d * d;
  }
  atomicAdd(&ssum, v);
  __syncthreads();
  if (threadIdx.x == 0)
    lq[row] += ldg[0] + ldg[1] + ldg[2] - 0.5f * (float)DD * LN2PI - ssum;
}

__global__ void zero_kernel(float* __restrict__ p, int n) {
  const int i = blockIdx.x * blockDim.x + threadIdx.x;
  if (i < n) p[i] = 0.f;
}

// ------------------------- host orchestration ------------------------------
static inline char* bump(char*& c, size_t bytes) {
  char* r = c;
  c += (bytes + 255) & ~(size_t)255;
  return r;
}

extern "C" void kernel_launch(void* const* d_in, const int* in_sizes, int n_in,
                              void* d_out, int out_size, void* d_ws, size_t ws_size,
                              hipStream_t stream) {
  (void)in_sizes; (void)n_in; (void)out_size; (void)ws_size;
  const float* x     = (const float*)d_in[0];
  const float* loc   = (const float*)d_in[1];
  const float* lsc   = (const float*)d_in[2];
  const float* Wi    = (const float*)d_in[3];
  const float* bi    = (const float*)d_in[4];
  const float* Wb    = (const float*)d_in[5];
  const float* bb    = (const float*)d_in[6];
  const float* Wo    = (const float*)d_in[7];
  const float* bo    = (const float*)d_in[8];
  const float* uw    = (const float*)d_in[9];
  const float* uh    = (const float*)d_in[10];
  const float* ud    = (const float*)d_in[11];
  const float* lu_lo = (const float*)d_in[12];
  const float* lu_up = (const float*)d_in[13];
  const float* lu_ud = (const float*)d_in[14];
  const float* lu_b  = (const float*)d_in[15];
  const int*   perms = (const int*)d_in[16];
  float* out = (float*)d_out;

  char* wc = (char*)d_ws;
  float* zCur   = (float*)bump(wc, (size_t)BROWS * DD * 4);
  float* zLin   = (float*)bump(wc, (size_t)BROWS * DKP * 4);
  float* ident  = (float*)bump(wc, (size_t)BROWS * FF * 4);
  float* pbuf   = (float*)bump(wc, (size_t)CHUNK * OUTP * 4);
  float* tf     = (float*)bump(wc, (size_t)CHUNK * HIDD * 4);
  float* r2f    = (float*)bump(wc, (size_t)CHUNK * HIDD * 4);
  u16*   zPermB = (u16*)bump(wc, (size_t)BROWS * DKP * 2);
  u16*   identB = (u16*)bump(wc, (size_t)BROWS * FKP * 2);
  u16*   Mt     = (u16*)bump(wc, (size_t)DKP * DKP * 2);
  u16*   WiT    = (u16*)bump(wc, (size_t)HIDD * FKP * 2);
  u16*   WbT    = (u16*)bump(wc, (size_t)4 * HIDD * HIDD * 2);
  u16*   WoT    = (u16*)bump(wc, (size_t)OUTP * HIDD * 2);
  u16*   tBf    = (u16*)bump(wc, (size_t)CHUNK * HIDD * 2);
  u16*   rBf    = (u16*)bump(wc, (size_t)CHUNK * HIDD * 2);
  u16*   r2Bf   = (u16*)bump(wc, (size_t)CHUNK * HIDD * 2);
  u16*   t2Bf   = (u16*)bump(wc, (size_t)CHUNK * HIDD * 2);
  float* knots  = (float*)bump(wc, (size_t)FF * 43 * 4);
  float* ldg    = (float*)bump(wc, 32);

  zero_kernel<<<(BROWS + 255) / 256, 256, 0, stream>>>(out, BROWS);

  for (int it = 0; it < 3; ++it) {
    const int i = 2 - it;
    const float* src = (it == 0) ? x : zCur;

    // --- LU stage ---
    build_Mt<<<(DKP * DKP + 255) / 256, 256, 0, stream>>>(
        lu_lo + (size_t)i * DD * DD, lu_up + (size_t)i * DD * DD, lu_ud + i * DD, Mt);
    ldiag_kernel<<<1, 256, 0, stream>>>(lu_ud + i * DD, ldg + i);
    gather_perm_bf<<<(BROWS * DKP + 255) / 256, 256, 0, stream>>>(
        src, perms + i * DD, zPermB);
    {
      dim3 g(DKP / 32, BROWS / 16);
      gemm_bf16_wmma<<<g, 32, 0, stream>>>(zPermB, Mt, lu_b + i * DD, nullptr,
                                           zLin, nullptr, DKP, DD, DKP, 0);
    }

    // --- unconditional spline ---
    build_uncond_knots<<<(FF + 63) / 64, 64, 0, stream>>>(
        uw + (size_t)i * FF * 8, uh + (size_t)i * FF * 8, ud + (size_t)i * FF * 7, knots);
    uncond_spline<<<BROWS, FKP, 0, stream>>>(zLin, knots, ident, identB, out);

    // --- layer weights -> transposed bf16 (L2 resident) ---
    conv_w_t<<<(HIDD * FKP + 255) / 256, 256, 0, stream>>>(
        Wi + (size_t)i * FF * HIDD, WiT, FF, HIDD, FKP, HIDD);
    for (int j = 0; j < 4; ++j)
      conv_w_t<<<(HIDD * HIDD + 255) / 256, 256, 0, stream>>>(
          Wb + ((size_t)i * 4 + j) * HIDD * HIDD, WbT + (size_t)j * HIDD * HIDD,
          HIDD, HIDD, HIDD, HIDD);
    conv_w_t<<<(OUTP * HIDD + 255) / 256, 256, 0, stream>>>(
        Wo + (size_t)i * HIDD * OUTW, WoT, HIDD, OUTW, HIDD, OUTP);

    const float* bi_i = bi + (size_t)i * HIDD;
    const float* bb_i = bb + (size_t)i * 4 * HIDD;
    const float* bo_i = bo + (size_t)i * OUTW;

    // --- resnet conditioner + conditional spline, in 1024-row chunks ---
    for (int c = 0; c < 4; ++c) {
      const u16* idc = identB + (size_t)c * CHUNK * FKP;
      dim3 gH(HIDD / 32, CHUNK / 16);
      // t = ident@Wi+bi            (f32 kept as residual; bf16 copy = relu(t))
      gemm_bf16_wmma<<<gH, 32, 0, stream>>>(idc, WiT, bi_i, nullptr,
                                            tf, tBf, FKP, HIDD, HIDD, 1);
      // r = relu(t)@Wb0+bb0        (only relu(r) bf16 needed)
      gemm_bf16_wmma<<<gH, 32, 0, stream>>>(tBf, WbT + 0 * HIDD * HIDD, bb_i + 0 * HIDD,
                                            nullptr, nullptr, rBf, HIDD, HIDD, HIDD, 1);
      // t2 = relu(r)@Wb1+bb1 + t   (f32 residual; bf16 copy = relu(t2))
      gemm_bf16_wmma<<<gH, 32, 0, stream>>>(rBf, WbT + 1 * HIDD * HIDD, bb_i + 1 * HIDD,
                                            tf, r2f, r2Bf, HIDD, HIDD, HIDD, 1);
      // r = relu(t2)@Wb2+bb2
      gemm_bf16_wmma<<<gH, 32, 0, stream>>>(r2Bf, WbT + 2 * HIDD * HIDD, bb_i + 2 * HIDD,
                                            nullptr, nullptr, rBf, HIDD, HIDD, HIDD, 1);
      // t3 = relu(r)@Wb3+bb3 + t2  (bf16, no relu: feeds Wo directly)
      gemm_bf16_wmma<<<gH, 32, 0, stream>>>(rBf, WbT + 3 * HIDD * HIDD, bb_i + 3 * HIDD,
                                            r2f, nullptr, t2Bf, HIDD, HIDD, HIDD, 0);
      // p = t3@Wo+bo
      dim3 gO(OUTP / 32, CHUNK / 16);
      gemm_bf16_wmma<<<gO, 32, 0, stream>>>(t2Bf, WoT, bo_i, nullptr,
                                            pbuf, nullptr, HIDD, OUTW, OUTP, 0);
      cond_spline<<<CHUNK, FKP, 0, stream>>>(zLin, ident, pbuf, zCur, out, c * CHUNK);
    }
  }
  final_kernel<<<BROWS, 256, 0, stream>>>(zCur, loc, lsc, ldg, out);
}
